// Pooler_29695403885261
// MI455X (gfx1250) — matrix operand order repaired and checked by reading, not compiled, not run on hardware
//
#include <hip/hip_runtime.h>
#include <hip/hip_bf16.h>
#include <math.h>
#include <stdint.h>

typedef __bf16 bf16_t;
typedef __attribute__((ext_vector_type(16))) __bf16 v16bf;
typedef __attribute__((ext_vector_type(8)))  __bf16 v8bf;
typedef __attribute__((ext_vector_type(8)))  float  v8f;

#define T0 64
#define B1 1024
#define HH 768
#define HD 384
#define GG 1536   // 4*HD
#define LL 9
#define LP 16     // padded label dim

// ---------------------------------------------------------------------------
// Fragment helpers (contiguous-k layouts used by this code, wave32)
// A 16x32 bf16: lane m = l&15; lanes 0-15 hold K 0..15, lanes 16-31
// hold K 16..31 -> two contiguous 8-elem chunks at k0+h*16 and +8.
// B 32x16 bf16 (stored N-major, ldb over K): lane n = l&15; lanes 0-15 K=0..15,
// lanes 16-31 K=16..31 -> 16 contiguous elems at k0 + h*16.
// D 16x16 f32: elem r -> (m = r + 8*(l>=16), n = l&15).
// ---------------------------------------------------------------------------
__device__ __forceinline__ v16bf load_a16x32(const bf16_t* A, int lda, int row0, int k0, int lane) {
    int m = lane & 15, h = lane >> 4;
    const bf16_t* p = A + (size_t)(row0 + m) * (size_t)lda + (size_t)(k0 + h * 16);
    v8bf lo = *(const v8bf*)p;
    v8bf hi = *(const v8bf*)(p + 8);
    return __builtin_shufflevector(lo, hi, 0,1,2,3,4,5,6,7,8,9,10,11,12,13,14,15);
}

__device__ __forceinline__ v16bf load_b32x16(const bf16_t* W, int ldb, int n0, int k0, int lane) {
    int n = lane & 15, h = lane >> 4;
    const bf16_t* p = W + (size_t)(n0 + n) * (size_t)ldb + (size_t)(k0 + h * 16);
    return *(const v16bf*)p;
}

__device__ __forceinline__ v8f wmma_bf16(v16bf a, v16bf b, v8f c) {
    return __builtin_amdgcn_wmma_f32_16x16x32_bf16(false, a, false, b, (short)0, c, false, false);
}

__device__ __forceinline__ float sigmoidf_(float x) { return 1.0f / (1.0f + expf(-x)); }

// Async global -> LDS 16-byte copy, tracked by ASYNCcnt (CDNA5 §10 / §15.18).
__device__ __forceinline__ void async_copy_b128(uint32_t lds_off, const bf16_t* gptr) {
    asm volatile("global_load_async_to_lds_b128 %0, %1, off"
                 :: "v"(lds_off), "v"(gptr) : "memory");
}
__device__ __forceinline__ void wait_asynccnt0() {
    asm volatile("s_wait_asynccnt 0x0" ::: "memory");
}

// ---------------------------------------------------------------------------
// f32 -> bf16 convert
// ---------------------------------------------------------------------------
__global__ void k_convert_bf16(const float* __restrict__ src, bf16_t* __restrict__ dst, int n) {
    int i = (blockIdx.x * blockDim.x + threadIdx.x) * 4;
    if (i + 3 < n) {
        dst[i+0] = (bf16_t)src[i+0];
        dst[i+1] = (bf16_t)src[i+1];
        dst[i+2] = (bf16_t)src[i+2];
        dst[i+3] = (bf16_t)src[i+3];
    } else {
        for (int j = i; j < n; ++j) dst[j] = (bf16_t)src[j];
    }
}

// Pad W_out (9,768) -> (16,768) bf16 with zero rows
__global__ void k_pad_wout(const float* __restrict__ wout, bf16_t* __restrict__ dst) {
    int i = blockIdx.x * blockDim.x + threadIdx.x;   // over 16*768
    if (i >= LP * HH) return;
    int l = i / HH;
    dst[i] = (l < LL) ? (bf16_t)wout[i] : (bf16_t)0.0f;
}

// ---------------------------------------------------------------------------
// xg GEMM: (T*B,768) x bf16 W^T -> (T*B, 3072) bf16, fused fwd(0:1536)/bwd(1536:3072)
// Block = 8 waves (2 m x 4 n), 128x256 macro tile, wave tile 64x64.
// Double-buffered async global->LDS staging of the 128x32 A panel and the
// 256x32 B panel per K-chunk; WMMA fragments gathered from LDS.
// grid (3072/256=12, 65536/128=512)
// ---------------------------------------------------------------------------
__global__ __launch_bounds__(256) void k_gemm_xg(
    const bf16_t* __restrict__ xbf,     // (T*B, 768)
    const bf16_t* __restrict__ Wf,      // (1536, 768) n-major
    const bf16_t* __restrict__ Wb,
    const float* __restrict__ bihf, const float* __restrict__ bhhf,
    const float* __restrict__ bihb, const float* __restrict__ bhhb,
    bf16_t* __restrict__ xg)            // (T*B, 3072)
{
    __shared__ __attribute__((aligned(32))) bf16_t Abuf[2][128][32];  // 16 KB
    __shared__ __attribute__((aligned(32))) bf16_t Bbuf[2][256][32];  // 32 KB

    int lane = threadIdx.x & 31;
    int warp = threadIdx.x >> 5;
    int m0b = blockIdx.y * 128;                     // block row base
    int dir = (blockIdx.x >= (GG / 256)) ? 1 : 0;   // 6 blocks per direction
    int n0b = blockIdx.x * 256 - dir * GG;          // block col base within direction
    const bf16_t* W = dir ? Wb : Wf;
    const float* bi = dir ? bihb : bihf;
    const float* bh = dir ? bhhb : bhhf;

    int mw = (warp >> 2) * 64;                      // wave row offset in block tile
    int nw = (warp & 3) * 64;                       // wave col offset in block tile

    // stage one 32-deep K chunk into LDS buffer `buf` (async, per-lane b128)
    auto stage = [&](int kk, int buf) {
        int t = threadIdx.x;
        #pragma unroll
        for (int i = 0; i < 2; ++i) {               // A: 128*32*2B = 512 x 16B
            int c = t + i * 256;
            int row = c >> 2;
            int kc  = (c & 3) * 8;
            async_copy_b128((uint32_t)(size_t)&Abuf[buf][row][kc],
                            xbf + (size_t)(m0b + row) * HH + kk + kc);
        }
        #pragma unroll
        for (int i = 0; i < 4; ++i) {               // B: 256*32*2B = 1024 x 16B
            int c = t + i * 256;
            int nrow = c >> 2;
            int kc   = (c & 3) * 8;
            async_copy_b128((uint32_t)(size_t)&Bbuf[buf][nrow][kc],
                            W + (size_t)(n0b + nrow) * HH + kk + kc);
        }
    };

    v8f acc[4][4];
    #pragma unroll
    for (int mi = 0; mi < 4; ++mi)
        #pragma unroll
        for (int ni = 0; ni < 4; ++ni) acc[mi][ni] = (v8f){};

    stage(0, 0);
    wait_asynccnt0();
    __syncthreads();

    for (int kk = 0; kk < HH; kk += 32) {
        int cur = (kk >> 5) & 1;
        if (kk + 32 < HH) stage(kk + 32, cur ^ 1);

        // gather fragments from LDS
        v16bf af[4], bfv[4];
        int m = lane & 15, h = lane >> 4;
        #pragma unroll
        for (int mi = 0; mi < 4; ++mi) {
            const v8bf* lo = (const v8bf*)&Abuf[cur][mw + mi * 16 + m][h * 16];
            const v8bf* hi = (const v8bf*)&Abuf[cur][mw + mi * 16 + m][h * 16 + 8];
            af[mi] = __builtin_shufflevector(*lo, *hi, 0,1,2,3,4,5,6,7,8,9,10,11,12,13,14,15);
        }
        #pragma unroll
        for (int ni = 0; ni < 4; ++ni)
            bfv[ni] = *(const v16bf*)&Bbuf[cur][nw + ni * 16 + m][h * 16];

        #pragma unroll
        for (int mi = 0; mi < 4; ++mi)
            #pragma unroll
            for (int ni = 0; ni < 4; ++ni)
                acc[mi][ni] = wmma_bf16(af[mi], bfv[ni], acc[mi][ni]);

        wait_asynccnt0();
        __syncthreads();
    }

    int n = lane & 15, h = lane >> 4;
    #pragma unroll
    for (int ni = 0; ni < 4; ++ni) {
        int col = n0b + nw + ni * 16 + n;
        float bias = bi[col] + bh[col];
        #pragma unroll
        for (int mi = 0; mi < 4; ++mi) {
            #pragma unroll
            for (int r = 0; r < 8; ++r) {
                int row = m0b + mw + mi * 16 + r + h * 8;
                xg[(size_t)row * 3072 + (size_t)dir * GG + col] = (bf16_t)(acc[mi][ni][r] + bias);
            }
        }
    }
}

// ---------------------------------------------------------------------------
// One LSTM time step, both directions (blockIdx.z). Each wave: 16 batch rows x
// 16 hidden cols, all 4 gate columns fused -> c,h in registers.
// grid (24/8=3, 1024/16=64, 2), block 256
// ---------------------------------------------------------------------------
__global__ __launch_bounds__(256) void k_lstm_step(
    const bf16_t* __restrict__ xg,      // (T*B, 3072)
    const bf16_t* __restrict__ Whhf,    // (1536, 384) n-major
    const bf16_t* __restrict__ Whhb,
    bf16_t* __restrict__ hf_all,        // (T, B, HD)
    bf16_t* __restrict__ hb_all,
    float* __restrict__ cf,             // (B, HD)
    float* __restrict__ cb,
    int s)                              // step 0..T-1
{
    int lane = threadIdx.x & 31;
    int warp = threadIdx.x >> 5;
    int dir  = blockIdx.z;
    int j0 = (blockIdx.x * 8 + warp) * 16;
    if (j0 >= HD) return;
    int m0 = blockIdx.y * 16;

    int t_out  = dir ? (T0 - 1 - s) : s;
    int t_prev = dir ? (T0 - s)     : (s - 1);
    const bf16_t* W = dir ? Whhb : Whhf;
    bf16_t* hall = dir ? hb_all : hf_all;
    float* c     = dir ? cb : cf;

    v8f acc[4];
    #pragma unroll
    for (int q = 0; q < 4; ++q) acc[q] = (v8f){};

    if (s > 0) {
        const bf16_t* hprev = hall + (size_t)t_prev * B1 * HD;
        for (int kk = 0; kk < HD; kk += 32) {
            v16bf a = load_a16x32(hprev, HD, m0, kk, lane);
            #pragma unroll
            for (int q = 0; q < 4; ++q) {
                v16bf b = load_b32x16(W, HD, q * HD + j0, kk, lane);
                acc[q] = wmma_bf16(a, b, acc[q]);
            }
        }
    }

    int n = lane & 15, h = lane >> 4;
    int j = j0 + n;
    #pragma unroll
    for (int r = 0; r < 8; ++r) {
        int b = m0 + r + h * 8;
        size_t xrow = ((size_t)t_out * B1 + b) * 3072 + (size_t)dir * GG;
        float gi = acc[0][r] + (float)xg[xrow + 0 * HD + j];
        float gf = acc[1][r] + (float)xg[xrow + 1 * HD + j];
        float gg = acc[2][r] + (float)xg[xrow + 2 * HD + j];
        float go = acc[3][r] + (float)xg[xrow + 3 * HD + j];
        size_t cidx = (size_t)b * HD + j;
        float cold = (s == 0) ? 0.0f : c[cidx];
        float cnew = sigmoidf_(gf) * cold + sigmoidf_(gi) * tanhf(gg);
        float hv   = sigmoidf_(go) * tanhf(cnew);
        c[cidx] = cnew;
        hall[((size_t)t_out * B1 + b) * HD + j] = (bf16_t)hv;
    }
}

// ---------------------------------------------------------------------------
// Emission GEMM: [h_f | h_b] (65536 x 768) x W_out^T -> em (65536 x 16) f32
// wave: 4 m-tiles x 1 n-tile; grid 65536/(8*64)=128, block 256
// ---------------------------------------------------------------------------
__global__ __launch_bounds__(256) void k_gemm_em(
    const bf16_t* __restrict__ hf,      // (T*B, 384)
    const bf16_t* __restrict__ hb,      // (T*B, 384)
    const bf16_t* __restrict__ WoutP,   // (16, 768) n-major, rows 9..15 zero
    const float* __restrict__ bout,     // (9,)
    float* __restrict__ em)             // (T*B, 16)
{
    int lane = threadIdx.x & 31;
    int warp = threadIdx.x >> 5;
    int m0 = (blockIdx.x * 8 + warp) * 64;

    v8f acc[4];
    #pragma unroll
    for (int mi = 0; mi < 4; ++mi) acc[mi] = (v8f){};

    for (int kk = 0; kk < HH; kk += 32) {
        const bf16_t* A = (kk < HD) ? hf : hb;
        int ka = (kk < HD) ? kk : (kk - HD);
        v16bf b = load_b32x16(WoutP, HH, 0, kk, lane);
        #pragma unroll
        for (int mi = 0; mi < 4; ++mi) {
            v16bf a = load_a16x32(A, HD, m0 + mi * 16, ka, lane);
            acc[mi] = wmma_bf16(a, b, acc[mi]);
        }
    }

    int n = lane & 15, h = lane >> 4;
    float bias = (n < LL) ? bout[n] : 0.0f;
    #pragma unroll
    for (int mi = 0; mi < 4; ++mi)
        #pragma unroll
        for (int r = 0; r < 8; ++r) {
            int row = m0 + mi * 16 + r + h * 8;
            em[(size_t)row * LP + n] = acc[mi][r] + bias;
        }
}

// ---------------------------------------------------------------------------
// CRF NLL terms: one thread per batch item; trans/start/end in LDS
// ---------------------------------------------------------------------------
__global__ __launch_bounds__(256) void k_crf(
    const float* __restrict__ em,       // (T*B, 16)
    const int* __restrict__ labels,     // (T, B)
    const float* __restrict__ start_t,  // (9,)
    const float* __restrict__ end_t,    // (9,)
    const float* __restrict__ trans,    // (9,9)
    float* __restrict__ terms)          // (B,) = num - Z
{
    __shared__ float s_tr[LL * LL];
    __shared__ float s_st[LL];
    __shared__ float s_en[LL];
    for (int i = threadIdx.x; i < LL * LL; i += blockDim.x) s_tr[i] = trans[i];
    if (threadIdx.x < LL) { s_st[threadIdx.x] = start_t[threadIdx.x]; s_en[threadIdx.x] = end_t[threadIdx.x]; }
    __syncthreads();

    int b = blockIdx.x * blockDim.x + threadIdx.x;
    if (b >= B1) return;

    // numerator
    int tprev = labels[0 * B1 + b];
    float num = s_st[tprev] + em[(size_t)(0 * B1 + b) * LP + tprev];
    for (int t = 1; t < T0; ++t) {
        int tag = labels[t * B1 + b];
        num += s_tr[tprev * LL + tag] + em[((size_t)t * B1 + b) * LP + tag];
        tprev = tag;
    }
    num += s_en[tprev];

    // partition function
    float alpha[LL];
    #pragma unroll
    for (int j = 0; j < LL; ++j) alpha[j] = s_st[j] + em[(size_t)(0 * B1 + b) * LP + j];
    for (int t = 1; t < T0; ++t) {
        float na[LL];
        const float* emt = em + ((size_t)t * B1 + b) * LP;
        #pragma unroll
        for (int j = 0; j < LL; ++j) {
            float mx = -3.0e38f;
            #pragma unroll
            for (int i = 0; i < LL; ++i) mx = fmaxf(mx, alpha[i] + s_tr[i * LL + j]);
            float ss = 0.0f;
            #pragma unroll
            for (int i = 0; i < LL; ++i) ss += expf(alpha[i] + s_tr[i * LL + j] - mx);
            na[j] = mx + logf(ss) + emt[j];
        }
        #pragma unroll
        for (int j = 0; j < LL; ++j) alpha[j] = na[j];
    }
    float mx = -3.0e38f;
    #pragma unroll
    for (int j = 0; j < LL; ++j) mx = fmaxf(mx, alpha[j] + s_en[j]);
    float ss = 0.0f;
    #pragma unroll
    for (int j = 0; j < LL; ++j) ss += expf(alpha[j] + s_en[j] - mx);
    float Z = mx + logf(ss);

    terms[b] = num - Z;
}

// final reduction: -mean over 1024 terms -> d_out[0]
__global__ __launch_bounds__(256) void k_reduce(const float* __restrict__ terms, float* __restrict__ out) {
    __shared__ float sm[256];
    float v = 0.0f;
    for (int i = threadIdx.x; i < B1; i += 256) v += terms[i];
    sm[threadIdx.x] = v;
    __syncthreads();
    for (int s = 128; s > 0; s >>= 1) {
        if (threadIdx.x < s) sm[threadIdx.x] += sm[threadIdx.x + s];
        __syncthreads();
    }
    if (threadIdx.x == 0) out[0] = -sm[0] / (float)B1;
}

// ---------------------------------------------------------------------------
extern "C" void kernel_launch(void* const* d_in, const int* in_sizes, int n_in,
                              void* d_out, int out_size, void* d_ws, size_t ws_size,
                              hipStream_t stream) {
    const float* x      = (const float*)d_in[0];
    const int*   labels = (const int*)d_in[1];
    const float* W_ihf  = (const float*)d_in[2];
    const float* W_hhf  = (const float*)d_in[3];
    const float* b_ihf  = (const float*)d_in[4];
    const float* b_hhf  = (const float*)d_in[5];
    const float* W_ihb  = (const float*)d_in[6];
    const float* W_hhb  = (const float*)d_in[7];
    const float* b_ihb  = (const float*)d_in[8];
    const float* b_hhb  = (const float*)d_in[9];
    const float* W_out  = (const float*)d_in[10];
    const float* b_out  = (const float*)d_in[11];
    const float* st_tr  = (const float*)d_in[12];
    const float* en_tr  = (const float*)d_in[13];
    const float* trans  = (const float*)d_in[14];
    float* out = (float*)d_out;

    size_t off = 0;
    auto alloc = [&](size_t bytes) -> char* {
        char* p = (char*)d_ws + off;
        off += (bytes + 255) & ~(size_t)255;
        return p;
    };

    bf16_t* xbf   = (bf16_t*)alloc((size_t)T0 * B1 * HH * 2);
    bf16_t* xg    = (bf16_t*)alloc((size_t)T0 * B1 * 3072 * 2);
    bf16_t* Wfb   = (bf16_t*)alloc((size_t)GG * HH * 2);
    bf16_t* Wbb   = (bf16_t*)alloc((size_t)GG * HH * 2);
    bf16_t* Whhfb = (bf16_t*)alloc((size_t)GG * HD * 2);
    bf16_t* Whhbb = (bf16_t*)alloc((size_t)GG * HD * 2);
    bf16_t* WoutP = (bf16_t*)alloc((size_t)LP * HH * 2);
    bf16_t* hf    = (bf16_t*)alloc((size_t)T0 * B1 * HD * 2);
    bf16_t* hb    = (bf16_t*)alloc((size_t)T0 * B1 * HD * 2);
    float*  cf    = (float*)alloc((size_t)B1 * HD * 4);
    float*  cb    = (float*)alloc((size_t)B1 * HD * 4);
    float*  em    = (float*)alloc((size_t)T0 * B1 * LP * 4);
    float*  terms = (float*)alloc((size_t)B1 * 4);

    // 1) converts
    {
        int n = T0 * B1 * HH;
        k_convert_bf16<<<(n / 4 + 255) / 256, 256, 0, stream>>>(x, xbf, n);
        n = GG * HH;
        k_convert_bf16<<<(n / 4 + 255) / 256, 256, 0, stream>>>(W_ihf, Wfb, n);
        k_convert_bf16<<<(n / 4 + 255) / 256, 256, 0, stream>>>(W_ihb, Wbb, n);
        n = GG * HD;
        k_convert_bf16<<<(n / 4 + 255) / 256, 256, 0, stream>>>(W_hhf, Whhfb, n);
        k_convert_bf16<<<(n / 4 + 255) / 256, 256, 0, stream>>>(W_hhb, Whhbb, n);
        k_pad_wout<<<(LP * HH + 255) / 256, 256, 0, stream>>>(W_out, WoutP);
    }

    // 2) input-projection GEMM (both directions fused, async-LDS staged)
    {
        dim3 grid(3072 / 256, (T0 * B1) / 128);
        k_gemm_xg<<<grid, 256, 0, stream>>>(xbf, Wfb, Wbb, b_ihf, b_hhf, b_ihb, b_hhb, xg);
    }

    // 3) 64 recurrent steps, both directions per launch
    {
        dim3 grid(3, B1 / 16, 2);
        for (int s = 0; s < T0; ++s)
            k_lstm_step<<<grid, 256, 0, stream>>>(xg, Whhfb, Whhbb, hf, hb, cf, cb, s);
    }

    // 4) emissions
    k_gemm_em<<<(T0 * B1) / (8 * 64), 256, 0, stream>>>(hf, hb, WoutP, b_out, em);

    // 5) CRF terms + reduction
    k_crf<<<(B1 + 255) / 256, 256, 0, stream>>>(em, labels, st_tr, en_tr, trans, terms);
    k_reduce<<<1, 256, 0, stream>>>(terms, out);
}
